// getBoxInfoListForOneImage_24421184045193
// MI455X (gfx1250) — 8-bit matrix operand order repaired and checked by reading, compiled, not run on hardware
//
#include <hip/hip_runtime.h>

typedef __attribute__((ext_vector_type(8))) int v8i;

#define W_OUT    768
#define H_OUT    768
#define NB       128
#define TILES_X  48
#define TILES_Y  48

__global__ void init_scores_kernel(unsigned int* __restrict__ s) {
    if (threadIdx.x < NB) s[threadIdx.x] = 0u;
}

__global__ __launch_bounds__(256) void build_masks_kernel(
    const float4* __restrict__ bboxes,
    uint4* __restrict__ ymask,
    uint4* __restrict__ xmask)
{
    const int idx  = blockIdx.x * 256 + threadIdx.x;
    const bool isX = idx >= H_OUT;
    const int  p   = isX ? (idx - H_OUT) : idx;
    const float pc = (float)p * 2.0f + 1.0f;

    unsigned m[4];
#pragma unroll
    for (int w = 0; w < 4; ++w) {
        unsigned mm = 0u;
#pragma clang loop unroll(disable)
        for (int i = 0; i < 32; ++i) {
            const float4 bx = bboxes[w * 32 + i];
            const float lo = isX ? bx.x : bx.y;
            const float hc = isX ? bx.z : bx.w;
            const int vv = (int)(!isX) |
                           (int)((bx.z - bx.x) * (bx.w - bx.y) > 0.0f);
            const int in = (int)(pc >= lo) & (int)(pc <= hc) & vv;
            mm |= in ? (1u << i) : 0u;
        }
        m[w] = mm;
    }
    uint4* dst = isX ? xmask : ymask;
    dst[p] = make_uint4(m[0], m[1], m[2], m[3]);
}

__global__ __launch_bounds__(32) void owner_scores_wmma_kernel(
    const float* __restrict__ conf,
    const uint4* __restrict__ ymask,
    const uint4* __restrict__ xmask,
    unsigned int* __restrict__ scores_bits)
{
    const int lane = threadIdx.x;
    const int tile = blockIdx.x;
    const int x0 = (tile % TILES_X) * 16;
    const int y0 = (tile / TILES_X) * 16;
    const int hi = lane >> 4;
    const int mn = lane & 15;

    const int col  = x0 + mn;
    const int rowb = y0 + (hi ? 8 : 0);
    float cv[8];
#pragma unroll
    for (int j = 0; j < 8; ++j)
        cv[j] = __builtin_nontemporal_load(&conf[(rowb + j) * W_OUT + col]);
    const uint4 ym = ymask[y0 + mn];
    const uint4 xm = xmask[x0 + mn];

    const unsigned ymw[4] = {ym.x, ym.y, ym.z, ym.w};
    const unsigned xmw[4] = {xm.x, xm.y, xm.z, xm.w};
    const unsigned sh8  = (unsigned)(hi * 8);
    const unsigned sh16 = (unsigned)(hi * 8);
    const unsigned hadd = hi ? 0x08080808u : 0u;

    v8i c_cnt = {};
    v8i c_ids = {};
    const int tA[4] = {0, 4, 16, 20};
    const int tB[4] = {0, 4, 16, 20};

#pragma unroll
    for (int c = 0; c < 2; ++c) {
        const unsigned uA0 = ymw[2 * c + 0] >> sh8;
        const unsigned uA1 = ymw[2 * c + 1] >> sh8;
        const unsigned uB0 = xmw[2 * c + 0] >> sh16;
        const unsigned uB1 = xmw[2 * c + 1] >> sh16;
        v8i a, aw, bm;
#pragma unroll
        for (int d = 0; d < 8; ++d) {
            const unsigned srcA = (d < 4) ? uA0 : uA1;
            const unsigned srcB = (d < 4) ? uB0 : uB1;
            const unsigned ta = (srcA >> tA[d & 3]) & 0xFu;
            const unsigned tb = (srcB >> tB[d & 3]) & 0xFu;
            const unsigned sa = (ta * 0x00204081u) & 0x01010101u;
            const unsigned sb = (tb * 0x00204081u) & 0x01010101u;
            const unsigned kbase  = (unsigned)(c * 64 + ((d < 4) ? 0 : 32) + tA[d & 3]);
            const unsigned wconst = 0x03020100u + kbase * 0x01010101u;
            a[d]  = (int)sa;
            aw[d] = (int)((sa * 0xFFu) & (wconst + hadd));
            bm[d] = (int)sb;
        }
        c_cnt = __builtin_amdgcn_wmma_i32_16x16x64_iu8(false, a,  false, bm, c_cnt, false, false);
        c_ids = __builtin_amdgcn_wmma_i32_16x16x64_iu8(false, aw, false, bm, c_ids, false, false);
    }

#pragma unroll
    for (int j = 0; j < 8; ++j) {
        if (c_cnt[j] == 1) {
            atomicMax(&scores_bits[c_ids[j]], __float_as_uint(cv[j]));
        }
    }
}

extern "C" void kernel_launch(void* const* d_in, const int* in_sizes, int n_in,
                              void* d_out, int out_size, void* d_ws, size_t ws_size,
                              hipStream_t stream) {
    const float*  conf   = (const float*)d_in[0];
    const float4* bboxes = (const float4*)d_in[1];
    unsigned int* scores_bits = (unsigned int*)d_out;

    uint4* ymask = (uint4*)d_ws;
    uint4* xmask = (uint4*)((char*)d_ws + H_OUT * sizeof(uint4));

    init_scores_kernel<<<1, 128, 0, stream>>>(scores_bits);
    build_masks_kernel<<<(H_OUT + W_OUT) / 256, 256, 0, stream>>>(bboxes, ymask, xmask);
    owner_scores_wmma_kernel<<<TILES_X * TILES_Y, 32, 0, stream>>>(conf, ymask, xmask, scores_bits);
}
